// Model_55568286876238
// MI455X (gfx1250) — matrix operand order repaired and checked by reading, not compiled, not run on hardware
//
#include <hip/hip_runtime.h>

// ---- problem constants (match reference) ----
#define BB 4
#define LL 96
#define NN 32
#define PB (LL*NN)        // 3072 points per batch
#define PP (BB*PB)        // 12288 points total
#define DM 128            // d_model
#define DC 32
#define DT 32
#define DCOORD (DC+DT)    // 64
#define KNB 16            // K neighbors
#define HID 256
#define LN_EPS 1e-5f

typedef _Float16 half_t;
typedef _Float16 v16h __attribute__((ext_vector_type(16)));
typedef float    v8f  __attribute__((ext_vector_type(8)));

// ---------------------------------------------------------------------------
// WMMA fragment helpers (CDNA5 16x16x32 f16 layouts, CDNA5 ISA)
// A (16xK=32): lanes 0-15 row=lane, halves[0..7]=K0..7, [8..15]=K16..23
//              lanes 16-31 row=lane-16, halves[0..7]=K8..15, [8..15]=K24..31
// rp must point at row[kbase + 8*half_sel].
__device__ __forceinline__ v16h load_a16(const half_t* rp) {
  v16h a;
#pragma unroll
  for (int j = 0; j < 8; ++j) { a[j] = rp[j]; a[8 + j] = rp[16 + j]; }
  return a;
}
// B (K=32 x 16): lane holds col=lane&15; halves j<8 -> K = 8*half_sel + j, j>=8 -> K = 8 + 8*half_sel + j.
// Weights pre-packed so the 32 K-values of (chunk,col) are contiguous.
__device__ __forceinline__ v16h load_b16(const half_t* bp) {
  v16h b;
#pragma unroll
  for (int j = 0; j < 16; ++j) b[j] = bp[j];
  return b;
}
__device__ __forceinline__ v8f wmma_f16(v16h a, v16h b, v8f c) {
  return __builtin_amdgcn_wmma_f32_16x16x32_f16(false, a, false, b,
                                                (short)0, c, false, false);
}

// ---------------------------------------------------------------------------
// Pack a row-major f32 [kdim x ncol] weight into B-fragment layout f16:
// dst[((k/32)*ncol + n)*32 + 16*((k>>3)&1) + 8*((k>>4)&1) + (k&7)] = src[k*ncol + n]
__global__ void pack_b_kernel(const float* __restrict__ src,
                              half_t* __restrict__ dst, int kdim, int ncol) {
  int idx = blockIdx.x * blockDim.x + threadIdx.x;
  if (idx >= kdim * ncol) return;
  int k = idx / ncol;
  int n = idx - k * ncol;
  dst[(((size_t)(k >> 5) * ncol + n) << 5) + (((k >> 3) & 1) << 4) + (((k >> 4) & 1) << 3) + (k & 7)] = (half_t)src[idx];
}

// ---------------------------------------------------------------------------
// Embedding: per point q -> p (e_c|e_t, 64) and h = MLP(x, 1->256->128).
// One 64-thread block per point.
__global__ __launch_bounds__(64) void embed_kernel(
    const float* __restrict__ x, const float* __restrict__ xmark,
    const float* __restrict__ chan_emb,
    const float* __restrict__ te_w1, const float* __restrict__ te_b1,
    const float* __restrict__ te_w2, const float* __restrict__ te_b2,
    const float* __restrict__ ve_w1, const float* __restrict__ ve_b1,
    const float* __restrict__ ve_w2, const float* __restrict__ ve_b2,
    float* __restrict__ p_out, float* __restrict__ h_out,
    half_t* __restrict__ h16_out) {
  __shared__ float hv[2 * DM];  // 256 ve hidden
  __shared__ float ht[2 * DT];  // 64 te hidden
  int q = blockIdx.x;
  int tid = threadIdx.x;
  int n = q & (NN - 1);
  int bl = q >> 5;  // (b*L + l) since N == 32
  float xv = x[q];
  float tv = xmark[bl];
  for (int j = tid; j < 2 * DM; j += 64)
    hv[j] = fmaxf(fmaf(xv, ve_w1[j], ve_b1[j]), 0.f);
  if (tid < 2 * DT)
    ht[tid] = fmaxf(fmaf(tv, te_w1[tid], te_b1[tid]), 0.f);
  __syncthreads();
  for (int d = tid; d < DM; d += 64) {
    float s = ve_b2[d];
    for (int j = 0; j < 2 * DM; ++j) s = fmaf(hv[j], ve_w2[j * DM + d], s);
    h_out[(size_t)q * DM + d] = s;
    h16_out[(size_t)q * DM + d] = (half_t)s;
  }
  if (tid < DT) {
    float s = te_b2[tid];
    for (int j = 0; j < 2 * DT; ++j) s = fmaf(ht[j], te_w2[j * DT + tid], s);
    p_out[(size_t)q * DCOORD + DC + tid] = s;
  }
  if (tid < DC) p_out[(size_t)q * DCOORD + tid] = chan_emb[n * DC + tid];
}

// ---------------------------------------------------------------------------
// Brute-force per-batch KNN over 64-dim coords (W_C == W_T == 1 so the two
// weighted sub-distances sum to the full squared distance).
#define KT 128
__global__ __launch_bounds__(KT) void knn_kernel(const float* __restrict__ p,
                                                 int* __restrict__ nb) {
  __shared__ float tile[KT][DCOORD];
  int b = blockIdx.y;
  int gb = b * PB;
  int i = blockIdx.x * KT + threadIdx.x;  // query index within batch
  float qp[DCOORD];
#pragma unroll
  for (int c = 0; c < DCOORD; ++c) qp[c] = p[(size_t)(gb + i) * DCOORD + c];
  float bd[KNB];
  int bi[KNB];
#pragma unroll
  for (int k = 0; k < KNB; ++k) { bd[k] = 3.0e38f; bi[k] = 0; }
  for (int jt = 0; jt < PB / KT; ++jt) {
    __syncthreads();
    for (int e = threadIdx.x; e < KT * DCOORD; e += KT)
      tile[e / DCOORD][e % DCOORD] = p[(size_t)(gb + jt * KT) * DCOORD + e];
    __syncthreads();
    for (int jj = 0; jj < KT; ++jj) {
      float d = 0.f;
#pragma unroll
      for (int c = 0; c < DCOORD; ++c) {
        float df = qp[c] - tile[jj][c];
        d = fmaf(df, df, d);
      }
      int j = jt * KT + jj;
      if (d < bd[KNB - 1]) {
        int pos = KNB - 1;
        while (pos > 0 && bd[pos - 1] > d) {
          bd[pos] = bd[pos - 1]; bi[pos] = bi[pos - 1]; --pos;
        }
        bd[pos] = d; bi[pos] = j;
      }
    }
  }
#pragma unroll
  for (int k = 0; k < KNB; ++k) nb[(size_t)(gb + i) * KNB + k] = gb + bi[k];
}

// ---------------------------------------------------------------------------
// Per-layer attention: one wave = one point (16 neighbor rows == WMMA M).
//  F = [p_rel(64) | h_i(128) | h_j(128)]  (16x320 f16 in LDS)
//  rk : F @ rk_w1 (10 k-chunks) -> relu -> H; scores = H @ rk_w2 + mask
//  mm : F(cols 192..319, 0..63) @ mm_w1 (6 chunks) -> relu -> H (reused)
//  msg: H @ mm_w2 (8 chunks) -> softmax-weighted sum over K -> agg (f16)
#define FST 328  // F row stride (halves), padded
#define HST 264  // H row stride (halves), padded
__global__ __launch_bounds__(64) void attn_kernel(
    const float* __restrict__ p, const half_t* __restrict__ h16,
    const float* __restrict__ xmark, const int* __restrict__ nb,
    const half_t* __restrict__ rk_w1t, const float* __restrict__ rk_b1,
    const float* __restrict__ rk_w2, const float* __restrict__ rk_b2,
    const half_t* __restrict__ mm_w1t, const float* __restrict__ mm_b1,
    const half_t* __restrict__ mm_w2t, const float* __restrict__ mm_b2,
    half_t* __restrict__ agg16) {
  __shared__ half_t F[2][KNB][FST];
  __shared__ half_t H[2][KNB][HST];
  __shared__ int nbs[2][KNB];
  __shared__ float sc[2][KNB];
  __shared__ float swt[2][KNB];

  int tid = threadIdx.x;
  int w = tid >> 5;
  int lane = tid & 31;
  int row = lane & 15;
  int hs = lane >> 4;
  int q = blockIdx.x * 2 + w;

  if (lane < KNB) nbs[w][lane] = nb[(size_t)q * KNB + lane];
  __syncthreads();

  // Build feature rows.
  for (int e = lane; e < KNB * 320; e += 32) {
    int r = e / 320, c = e - r * 320;
    int g = nbs[w][r];
    half_t v;
    if (c < DCOORD)
      v = (half_t)(p[(size_t)q * DCOORD + c] - p[(size_t)g * DCOORD + c]);
    else if (c < DCOORD + DM)
      v = h16[(size_t)q * DM + (c - DCOORD)];
    else
      v = h16[(size_t)g * DM + (c - DCOORD - DM)];
    F[w][r][c] = v;
  }
  __syncthreads();

  // rk GEMM: 320 -> 256, relu
  for (int nt = 0; nt < 16; ++nt) {
    int col = nt * 16 + row;
    v8f acc = {};
#pragma unroll
    for (int kc = 0; kc < 10; ++kc) {
      v16h a = load_a16(&F[w][row][kc * 32 + 8 * hs]);
      v16h b = load_b16(&rk_w1t[(((size_t)kc * HID + col) << 5) + 16 * hs]);
      acc = wmma_f16(a, b, acc);
    }
    float bias = rk_b1[col];
#pragma unroll
    for (int r = 0; r < 8; ++r)
      H[w][8 * hs + r][col] = (half_t)fmaxf(acc[r] + bias, 0.f);
  }
  __syncthreads();

  // scores + causal mask + softmax over K
  {
    float ti = xmark[q >> 5];
    if (lane < KNB) {
      float s = rk_b2[0];
      for (int j = 0; j < HID; ++j)
        s = fmaf((float)H[w][lane][j], rk_w2[j], s);
      int g = nbs[w][lane];
      float tj = xmark[g >> 5];
      if (tj > ti) s = -3.0e38f;
      sc[w][lane] = s;
    }
    __syncthreads();
    if (lane < KNB) {
      float m = sc[w][0];
#pragma unroll
      for (int k = 1; k < KNB; ++k) m = fmaxf(m, sc[w][k]);
      float den = 0.f;
#pragma unroll
      for (int k = 0; k < KNB; ++k) den += expf(sc[w][k] - m);
      swt[w][lane] = expf(sc[w][lane] - m) / den;
    }
    __syncthreads();
  }

  // mm hidden GEMM: [h_j | p_rel] (192) -> 256, relu (H reused)
  for (int nt = 0; nt < 16; ++nt) {
    int col = nt * 16 + row;
    v8f acc = {};
#pragma unroll
    for (int kc = 0; kc < 6; ++kc) {
      int fb = (kc < 4) ? (192 + kc * 32) : ((kc - 4) * 32);
      v16h a = load_a16(&F[w][row][fb + 8 * hs]);
      v16h b = load_b16(&mm_w1t[(((size_t)kc * HID + col) << 5) + 16 * hs]);
      acc = wmma_f16(a, b, acc);
    }
    float bias = mm_b1[col];
#pragma unroll
    for (int r = 0; r < 8; ++r)
      H[w][8 * hs + r][col] = (half_t)fmaxf(acc[r] + bias, 0.f);
  }
  __syncthreads();

  // msg GEMM (256 -> 128) fused with softmax-weighted aggregation over K.
  for (int nt = 0; nt < 8; ++nt) {
    int col = nt * 16 + row;
    v8f acc = {};
#pragma unroll
    for (int kc = 0; kc < 8; ++kc) {
      v16h a = load_a16(&H[w][row][kc * 32 + 8 * hs]);
      v16h b = load_b16(&mm_w2t[(((size_t)kc * DM + col) << 5) + 16 * hs]);
      acc = wmma_f16(a, b, acc);
    }
    float part = 0.f;
#pragma unroll
    for (int r = 0; r < 8; ++r) part = fmaf(swt[w][8 * hs + r], acc[r], part);
    part += __shfl_xor(part, 16, 32);  // combine M 0-7 with M 8-15 halves
    if (lane < 16)
      agg16[(size_t)q * DM + col] = (half_t)(part + mm_b2[col]);  // sum(w)==1
  }
}

// ---------------------------------------------------------------------------
// update MLP (128->256->128) + residual + layernorm. One wave = 16 points.
__global__ __launch_bounds__(64) void um_kernel(
    const half_t* __restrict__ agg16, const float* __restrict__ h_in,
    const half_t* __restrict__ um_w1t, const float* __restrict__ um_b1,
    const half_t* __restrict__ um_w2t, const float* __restrict__ um_b2,
    const float* __restrict__ lng, const float* __restrict__ lnb,
    float* __restrict__ h_out, half_t* __restrict__ h16_out) {
  __shared__ half_t H[2][16][HST];
  __shared__ float Y[2][16][DM];
  int tid = threadIdx.x;
  int w = tid >> 5;
  int lane = tid & 31;
  int row = lane & 15;
  int hs = lane >> 4;
  int q0 = (blockIdx.x * 2 + w) * 16;

  // hidden GEMM from global f16 agg
  for (int nt = 0; nt < 16; ++nt) {
    int col = nt * 16 + row;
    v8f acc = {};
#pragma unroll
    for (int kc = 0; kc < 4; ++kc) {
      v16h a = load_a16(&agg16[(size_t)(q0 + row) * DM + kc * 32 + 8 * hs]);
      v16h b = load_b16(&um_w1t[(((size_t)kc * HID + col) << 5) + 16 * hs]);
      acc = wmma_f16(a, b, acc);
    }
    float bias = um_b1[col];
#pragma unroll
    for (int r = 0; r < 8; ++r)
      H[w][8 * hs + r][col] = (half_t)fmaxf(acc[r] + bias, 0.f);
  }
  __syncthreads();

  // output GEMM + bias + residual
  for (int nt = 0; nt < 8; ++nt) {
    int col = nt * 16 + row;
    v8f acc = {};
#pragma unroll
    for (int kc = 0; kc < 8; ++kc) {
      v16h a = load_a16(&H[w][row][kc * 32 + 8 * hs]);
      v16h b = load_b16(&um_w2t[(((size_t)kc * DM + col) << 5) + 16 * hs]);
      acc = wmma_f16(a, b, acc);
    }
    float bias = um_b2[col];
#pragma unroll
    for (int r = 0; r < 8; ++r) {
      int m = 8 * hs + r;
      Y[w][m][col] = acc[r] + bias + h_in[(size_t)(q0 + m) * DM + col];
    }
  }
  __syncthreads();

  // layernorm per point
  if (lane < 16) {
    int pq = q0 + lane;
    float mu = 0.f;
    for (int d = 0; d < DM; ++d) mu += Y[w][lane][d];
    mu *= (1.f / DM);
    float var = 0.f;
    for (int d = 0; d < DM; ++d) {
      float df = Y[w][lane][d] - mu;
      var = fmaf(df, df, var);
    }
    var *= (1.f / DM);
    float inv = rsqrtf(var + LN_EPS);
    for (int d = 0; d < DM; ++d) {
      float v = fmaf(lng[d], (Y[w][lane][d] - mu) * inv, lnb[d]);
      h_out[(size_t)pq * DM + d] = v;
      h16_out[(size_t)pq * DM + d] = (half_t)v;
    }
  }
}

// ---------------------------------------------------------------------------
extern "C" void kernel_launch(void* const* d_in, const int* in_sizes, int n_in,
                              void* d_out, int out_size, void* d_ws,
                              size_t ws_size, hipStream_t stream) {
  const float* x        = (const float*)d_in[0];
  const float* xmark    = (const float*)d_in[1];
  const float* chan_emb = (const float*)d_in[2];
  const float* te_w1 = (const float*)d_in[3];
  const float* te_b1 = (const float*)d_in[4];
  const float* te_w2 = (const float*)d_in[5];
  const float* te_b2 = (const float*)d_in[6];
  const float* ve_w1 = (const float*)d_in[7];
  const float* ve_b1 = (const float*)d_in[8];
  const float* ve_w2 = (const float*)d_in[9];
  const float* ve_b2 = (const float*)d_in[10];
  const float* rk_w1 = (const float*)d_in[11];
  const float* rk_b1 = (const float*)d_in[12];
  const float* rk_w2 = (const float*)d_in[13];
  const float* rk_b2 = (const float*)d_in[14];
  const float* mm_w1 = (const float*)d_in[15];
  const float* mm_b1 = (const float*)d_in[16];
  const float* mm_w2 = (const float*)d_in[17];
  const float* mm_b2 = (const float*)d_in[18];
  const float* um_w1 = (const float*)d_in[19];
  const float* um_b1 = (const float*)d_in[20];
  const float* um_w2 = (const float*)d_in[21];
  const float* um_b2 = (const float*)d_in[22];
  const float* ln_g  = (const float*)d_in[23];
  const float* ln_b  = (const float*)d_in[24];
  (void)in_sizes; (void)n_in; (void)out_size; (void)ws_size;

  char* base = (char*)d_ws;
  size_t off = 0;
  auto carve = [&](size_t bytes) -> char* {
    char* ptr = base + off;
    off = (off + bytes + 255) & ~(size_t)255;
    return ptr;
  };
  float*  p_ws  = (float*)carve((size_t)PP * DCOORD * 4);
  float*  h0    = (float*)carve((size_t)PP * DM * 4);
  float*  h1    = (float*)carve((size_t)PP * DM * 4);
  half_t* h0f16 = (half_t*)carve((size_t)PP * DM * 2);
  half_t* h1f16 = (half_t*)carve((size_t)PP * DM * 2);
  half_t* agg16 = (half_t*)carve((size_t)PP * DM * 2);
  int*    nbws  = (int*)carve((size_t)PP * KNB * 4);
  half_t* rkw1t = (half_t*)carve((size_t)2 * 320 * HID * 2);
  half_t* mmw1t = (half_t*)carve((size_t)2 * 192 * HID * 2);
  half_t* mmw2t = (half_t*)carve((size_t)2 * HID * DM * 2);
  half_t* umw1t = (half_t*)carve((size_t)2 * DM * HID * 2);
  half_t* umw2t = (half_t*)carve((size_t)2 * HID * DM * 2);

  embed_kernel<<<PP, 64, 0, stream>>>(x, xmark, chan_emb, te_w1, te_b1, te_w2,
                                      te_b2, ve_w1, ve_b1, ve_w2, ve_b2, p_ws,
                                      h0, h0f16);
  for (int l = 0; l < 2; ++l) {
    auto packs = [&](const float* src, half_t* dst, int kd, int nc) {
      int el = kd * nc;
      pack_b_kernel<<<(el + 255) / 256, 256, 0, stream>>>(
          src + (size_t)l * el, dst + (size_t)l * el, kd, nc);
    };
    packs(rk_w1, rkw1t, 320, HID);
    packs(mm_w1, mmw1t, 192, HID);
    packs(mm_w2, mmw2t, HID, DM);
    packs(um_w1, umw1t, DM, HID);
    packs(um_w2, umw2t, HID, DM);
  }
  knn_kernel<<<dim3(PB / KT, BB), KT, 0, stream>>>(p_ws, nbws);

  const float* hin = h0;
  const half_t* hin16 = h0f16;
  for (int l = 0; l < 2; ++l) {
    attn_kernel<<<PP / 2, 64, 0, stream>>>(
        p_ws, hin16, xmark, nbws,
        rkw1t + (size_t)l * 320 * HID, rk_b1 + l * HID, rk_w2 + l * HID,
        rk_b2 + l,
        mmw1t + (size_t)l * 192 * HID, mm_b1 + l * HID,
        mmw2t + (size_t)l * HID * DM, mm_b2 + l * DM, agg16);
    float* hout = (l == 1) ? (float*)d_out : h1;
    half_t* hout16 = (l == 1) ? h0f16 : h1f16;  // scratch on final layer
    um_kernel<<<PP / 32, 64, 0, stream>>>(
        agg16, hin,
        umw1t + (size_t)l * DM * HID, um_b1 + l * HID,
        umw2t + (size_t)l * HID * DM, um_b2 + l * DM,
        ln_g + l * DM, ln_b + l * DM, hout, hout16);
    hin = hout;
    hin16 = hout16;
  }
}
